// self_LSTM_90907277787373
// MI455X (gfx1250) — matrix operand order repaired and checked by reading, not compiled, not run on hardware
//
#include <hip/hip_runtime.h>

typedef __attribute__((ext_vector_type(16))) __bf16 v16bf;
typedef __attribute__((ext_vector_type(8)))  __bf16 v8bf;
typedef __attribute__((ext_vector_type(8)))  float  v8f;
typedef int v4i __attribute__((vector_size(16)));

#define BSZ   128   // batch
#define TT    1024  // timesteps
#define HH    512   // hidden
#define CC    10    // classes
#define NG    8     // batch groups (16 rows each)
#define NWG   16    // workgroups per group
#define MB    16    // batch rows per group
#define JW    32    // hidden units per WG
#define HP    520   // padded LDS pitch (bf16 elems): 1040B = 65*16B, bank-conflict free

#if defined(__AMDGCN__) && __has_builtin(__builtin_amdgcn_global_load_async_to_lds_b128)
#define HAVE_ASYNC_LDS 1
#else
#define HAVE_ASYNC_LDS 0
#endif

#if HAVE_ASYNC_LDS
typedef __attribute__((address_space(1))) v4i* gas_v4i_ptr;
typedef __attribute__((address_space(3))) v4i* las_v4i_ptr;
#endif

__device__ __forceinline__ float sigm_f(float x) {
    return 1.0f / (1.0f + __expf(-x));
}
__device__ __forceinline__ float tanh_f(float x) {
    float e = __expf(2.0f * x);
    return (e - 1.0f) / (e + 1.0f);
}

// Issue the (async) copy of h_{t-1} tile (16x512 bf16) global -> LDS, plus x_t -> LDS.
__device__ __forceinline__ void stage_issue(int t, const __bf16* __restrict__ hbuf,
                                            const float* __restrict__ x,
                                            __bf16 (*h_lds)[HP], float* x_lds,
                                            int g, int b0, int tid) {
    const int rb = ((t & 1) ^ 1);            // buffer holding h_{t-1}
    const int sb = tid >> 4;                 // batch row this thread stages
    const int kb = (tid & 15) * 32;          // 32 bf16 = 64B contiguous chunk
    const __bf16* src = hbuf + (((size_t)rb * NG + g) * MB + sb) * HH + kb;
    __bf16* dst = &h_lds[sb][kb];
#if HAVE_ASYNC_LDS
    gas_v4i_ptr gp = (gas_v4i_ptr)(uintptr_t)src;
    las_v4i_ptr lp = (las_v4i_ptr)(uint32_t)(uintptr_t)dst;
    __builtin_amdgcn_global_load_async_to_lds_b128(gp, lp,  0, 0);
    __builtin_amdgcn_global_load_async_to_lds_b128(gp, lp, 16, 0);
    __builtin_amdgcn_global_load_async_to_lds_b128(gp, lp, 32, 0);
    __builtin_amdgcn_global_load_async_to_lds_b128(gp, lp, 48, 0);
#else
    const v8bf* s8 = (const v8bf*)src;
    v8bf* d8 = (v8bf*)dst;
    #pragma unroll
    for (int q = 0; q < 4; ++q) d8[q] = s8[q];
#endif
    if (tid < MB) x_lds[tid] = x[(size_t)(b0 + tid) * TT + t];
}

__device__ __forceinline__ void stage_wait() {
#if HAVE_ASYNC_LDS
 #if __has_builtin(__builtin_amdgcn_s_wait_asynccnt)
    __builtin_amdgcn_s_wait_asynccnt(0);
 #else
    asm volatile("s_wait_asynccnt 0" ::: "memory");
 #endif
#endif
}

// ---------------- setup: zero h ping-pong bufs + barriers, init out = b_fc --------------
__global__ void lstm_setup_kernel(float* __restrict__ out, const float* __restrict__ b_fc,
                                  __bf16* __restrict__ hbuf, unsigned* __restrict__ bar) {
    size_t i = (size_t)blockIdx.x * blockDim.x + threadIdx.x;
    size_t stride = (size_t)gridDim.x * blockDim.x;
    const size_t n_out  = (size_t)BSZ * TT * CC;       // 1,310,720
    const size_t n_hbuf = (size_t)2 * NG * MB * HH;    // 131,072
    for (size_t k = i; k < n_out; k += stride)  out[k] = b_fc[k % CC];
    for (size_t k = i; k < n_hbuf; k += stride) hbuf[k] = (__bf16)0.0f;
    for (size_t k = i; k < NG * 32; k += stride) bar[k] = 0u;
}

// ---------------- main persistent recurrence kernel ------------------------------------
__global__ __launch_bounds__(256, 1)
void lstm_main_kernel(const float* __restrict__ x,     // (B,T)
                      const float* __restrict__ w_ih,  // (4H,1)
                      const float* __restrict__ w_hh,  // (4H,H)
                      const float* __restrict__ b_ih,  // (4H)
                      const float* __restrict__ b_hh,  // (4H)
                      const float* __restrict__ w_t,   // (2H,1)
                      const float* __restrict__ w_fc,  // (C,H)
                      float* __restrict__ out,         // (B,T,C)
                      __bf16* __restrict__ hbuf,       // [2][NG][MB][H] bf16
                      float* __restrict__ aPart,       // [2][NG][NWG][MB]
                      float* __restrict__ dPart,       // [2][NG][NWG][MB]
                      unsigned* __restrict__ bar)      // [NG][32] {count, gen}
{
    const int g    = blockIdx.x / NWG;   // batch group
    const int wg   = blockIdx.x % NWG;   // workgroup within group
    const int tid  = threadIdx.x;
    const int lane = tid & 31;
    const int wave = tid >> 5;           // 0..7
    const int b0   = g * MB;             // batch base
    const int j0   = wg * JW;            // hidden-unit base owned by this WG

    __shared__ __bf16 h_lds[MB][HP];          // h_{t-1} tile, bf16
    __shared__ float  gates_lds[MB][132];     // 16 x 128 gate pre-activations
    __shared__ float  x_lds[MB];
    __shared__ float  a_lds[MB];
    __shared__ float  d_lds[MB];

    // ---- preload this wave's weight slice as WMMA B-fragments (kept in VGPRs) ----
    // wave computes gate columns [wave*16, wave*16+16) of this WG's 128 columns.
    // B 32x16 bf16, k order used here: lane = N (mod 16); lanes<16 hold K 0..15, lanes>=16 hold K 16..31.
    const int ncol = wave * 16 + (lane & 15);                  // 0..127 local gate col
    const int grow = (ncol >> 5) * HH + j0 + (ncol & 31);      // row of w_hh (4H x H)
    const float* wr = w_hh + (size_t)grow * HH;
    v16bf bfrag[16];
    #pragma unroll
    for (int kt = 0; kt < 16; ++kt) {
        const int k0 = kt * 32 + ((lane >> 4) << 4);
        v16bf v;
        #pragma unroll
        for (int e = 0; e < 16; ++e) v[e] = (__bf16)wr[k0 + e];
        bfrag[kt] = v;
    }
    const float wihv  = w_ih[grow];
    const float biasv = b_ih[grow] + b_hh[grow];

    // ---- per-thread elementwise state: 2 cells (b = tid/16, r = (tid%16)*2 + {0,1}) ----
    const int eb  = tid >> 4;          // 0..15 batch row in group
    const int er0 = (tid & 15) * 2;    // local hidden index
    float c_st[2] = {0.f, 0.f}, S1[2] = {0.f, 0.f}, S2[2] = {0.f, 0.f};
    const float w1v[2] = { w_t[j0 + er0], w_t[j0 + er0 + 1] };
    const float w2v[2] = { w_t[HH + j0 + er0], w_t[HH + j0 + er0 + 1] };
    float wfc[CC][2];
    #pragma unroll
    for (int c = 0; c < CC; ++c) {
        wfc[c][0] = w_fc[(size_t)c * HH + j0 + er0];
        wfc[c][1] = w_fc[(size_t)c * HH + j0 + er0 + 1];
    }

    unsigned* cntp = &bar[g * 32 + 0];
    unsigned* genp = &bar[g * 32 + 1];

    // software pipeline: issue staging for t=0 (reads zeroed h buffer) before the loop
    stage_issue(0, hbuf, x, h_lds, x_lds, g, b0, tid);

    for (int t = 0; t < TT; ++t) {
        const int wb = t & 1;         // buffer we write h_t into

        // ---- wait for staged h_{t-1} / x_t (issued at end of previous iteration) ----
        stage_wait();
        __syncthreads();

        // ---- GEMM: each wave computes a 16x16 tile of gates = h @ Wslice^T ----
        v8f acc = {0.f, 0.f, 0.f, 0.f, 0.f, 0.f, 0.f, 0.f};
        {
            const int m  = lane & 15;
            const int ko = (lane >> 4) ? 16 : 0;  // A 16x32 bf16 lane layout
            #pragma unroll
            for (int kt = 0; kt < 16; ++kt) {
                const v8bf* p = (const v8bf*)&h_lds[m][kt * 32 + ko];
                v8bf lo = p[0];   // K = base+0..7   (or +16..23 for upper lanes)
                v8bf hi = p[1];   // K = base+8..15  (or +24..31)
                v16bf afrag = __builtin_shufflevector(lo, hi,
                    0,1,2,3,4,5,6,7,8,9,10,11,12,13,14,15);
                acc = __builtin_amdgcn_wmma_f32_16x16x32_bf16(
                    false, afrag, false, bfrag[kt], (short)0, acc, false, false);
            }
        }

        // ---- add x*w_ih + bias; spill gate tile to LDS (C layout: elem v -> M=v+8*(lane/16), N=lane%16) ----
        {
            const int n = wave * 16 + (lane & 15);
            const int mo = (lane >> 4) << 3;
            #pragma unroll
            for (int v = 0; v < 8; ++v)
                gates_lds[v + mo][n] = acc[v] + x_lds[v + mo] * wihv + biasv;
        }
        __syncthreads();

        // ---- elementwise LSTM cell for 2 owned cells ----
        float hnew[2];
        float aP = 0.f, dP = 0.f;
        #pragma unroll
        for (int q = 0; q < 2; ++q) {
            const int r = er0 + q;
            const float ig = gates_lds[eb][r];
            const float fg = gates_lds[eb][32 + r];
            const float gg = gates_lds[eb][64 + r];
            const float og = gates_lds[eb][96 + r];
            const float cn = sigm_f(fg) * c_st[q] + sigm_f(ig) * tanh_f(gg);
            c_st[q] = cn;
            const float h = sigm_f(og) * tanh_f(cn);
            hnew[q] = h;
            aP += h * w1v[q];
            dP += h * w2v[q];
            hbuf[(((size_t)wb * NG + g) * MB + eb) * HH + j0 + r] = (__bf16)h;
        }
        // reduce partial dots over the 16 lanes sharing batch row eb
        #pragma unroll
        for (int msk = 1; msk < 16; msk <<= 1) {
            aP += __shfl_xor(aP, msk, 32);
            dP += __shfl_xor(dP, msk, 32);
        }
        if ((tid & 15) == 0) {
            const size_t pi = (((size_t)wb * NG + g) * NWG + wg) * MB + eb;
            aPart[pi] = aP;
            dPart[pi] = dP;
        }
        __threadfence();
        __syncthreads();

        // ---- inter-workgroup barrier for this group (release/acquire, generation count) ----
        if (tid == 0) {
            unsigned prev = __hip_atomic_fetch_add(cntp, 1u, __ATOMIC_ACQ_REL,
                                                   __HIP_MEMORY_SCOPE_AGENT);
            if (prev == NWG - 1) {
                __hip_atomic_store(cntp, 0u, __ATOMIC_RELAXED, __HIP_MEMORY_SCOPE_AGENT);
                __hip_atomic_fetch_add(genp, 1u, __ATOMIC_ACQ_REL, __HIP_MEMORY_SCOPE_AGENT);
            } else {
                while (__hip_atomic_load(genp, __ATOMIC_ACQUIRE,
                                         __HIP_MEMORY_SCOPE_AGENT) < (unsigned)(t + 1)) {
                    __builtin_amdgcn_s_sleep(1);
                }
            }
        }
        __syncthreads();

        // ---- h_t is now published: immediately issue async staging for step t+1, ----
        // ---- then hide its latency under the gather / attention / output work.    ----
        if (t + 1 < TT)
            stage_issue(t + 1, hbuf, x, h_lds, x_lds, g, b0, tid);

        // ---- gather full dot products a_t = h.w1, d_t = h.w2 across the group ----
        if (tid < MB) {
            const size_t base = ((size_t)wb * NG + g) * NWG * MB + tid;
            float sa = 0.f, sd = 0.f;
            #pragma unroll
            for (int w = 0; w < NWG; ++w) {
                sa += aPart[base + (size_t)w * MB];
                sd += dPart[base + (size_t)w * MB];
            }
            a_lds[tid] = sa;
            d_lds[tid] = sd;
        }
        __syncthreads();

        // ---- attention, output projection partials, state update ----
        {
            const float a_t = a_lds[eb];
            const float d_t = d_lds[eb];
            float red[CC];
            #pragma unroll
            for (int c = 0; c < CC; ++c) red[c] = 0.f;
            #pragma unroll
            for (int q = 0; q < 2; ++q) {
                const float vout = hnew[q] + a_t * S1[q] + S2[q];
                S1[q] += hnew[q];
                S2[q] += d_t * hnew[q];
                #pragma unroll
                for (int c = 0; c < CC; ++c) red[c] += vout * wfc[c][q];
            }
            #pragma unroll
            for (int c = 0; c < CC; ++c) {
                #pragma unroll
                for (int msk = 1; msk < 16; msk <<= 1)
                    red[c] += __shfl_xor(red[c], msk, 32);
            }
            if ((tid & 15) == 0) {
                float* op = out + ((size_t)(b0 + eb) * TT + t) * CC;
                #pragma unroll
                for (int c = 0; c < CC; ++c) atomicAdd(op + c, red[c]);
            }
        }
        // next iteration's stage_wait + __syncthreads separates a_lds/x_lds/h_lds reuse
    }
}

extern "C" void kernel_launch(void* const* d_in, const int* in_sizes, int n_in,
                              void* d_out, int out_size, void* d_ws, size_t ws_size,
                              hipStream_t stream) {
    const float* x    = (const float*)d_in[0];
    const float* w_ih = (const float*)d_in[1];
    const float* w_hh = (const float*)d_in[2];
    const float* b_ih = (const float*)d_in[3];
    const float* b_hh = (const float*)d_in[4];
    const float* w_t  = (const float*)d_in[5];
    const float* w_fc = (const float*)d_in[6];
    const float* b_fc = (const float*)d_in[7];
    float* out = (float*)d_out;

    // workspace carve-up
    char* ws = (char*)d_ws;
    __bf16* hbuf = (__bf16*)ws;                                // 2*NG*MB*H bf16 = 256 KB
    ws += (size_t)2 * NG * MB * HH * sizeof(__bf16);
    float* aPart = (float*)ws;  ws += (size_t)2 * NG * NWG * MB * sizeof(float);
    float* dPart = (float*)ws;  ws += (size_t)2 * NG * NWG * MB * sizeof(float);
    unsigned* bar = (unsigned*)ws;                             // NG * 32 uints

    lstm_setup_kernel<<<dim3(2048), dim3(256), 0, stream>>>(out, b_fc, hbuf, bar);
    lstm_main_kernel<<<dim3(NG * NWG), dim3(256), 0, stream>>>(
        x, w_ih, w_hh, b_ih, b_hh, w_t, w_fc, out, hbuf, aPart, dPart, bar);
}
